// OptimizedMultiHeadAttention_7773890805826
// MI455X (gfx1250) — matrix operand order repaired and checked by reading, not compiled, not run on hardware
//
#include <hip/hip_runtime.h>

// ---------------------------------------------------------------------------
// Types for CDNA5 WMMA (wave32)
// ---------------------------------------------------------------------------
typedef __attribute__((ext_vector_type(16))) __bf16 bf16x16;
typedef __attribute__((ext_vector_type(8)))  float  v8f;

#define DEV __device__ __forceinline__

// ---------------------------------------------------------------------------
// CDNA5 async global->LDS copy (ASYNCcnt-tracked Tensor/async path).
// LDS address operand = low 32 bits of generic pointer.
// ---------------------------------------------------------------------------
DEV void async_copy_b128(void* lds_ptr, const void* gptr) {
    unsigned lds = (unsigned)(unsigned long long)lds_ptr;
    unsigned long long ga = (unsigned long long)gptr;
    asm volatile("global_load_async_to_lds_b128 %0, %1, off"
                 :: "v"(lds), "v"(ga) : "memory");
}
DEV void wait_async_all() {
    asm volatile("s_wait_asynccnt 0x0" ::: "memory");
}

// ---------------------------------------------------------------------------
// Fragment loaders (layouts per CDNA5 ISA 7.12.2)
// A: 16x32 bf16. lanes 0-15: M=lane, VGPR0..3 K={0..7} pairs, VGPR4..7 K={16..23}
//    lanes16-31: M=lane-16, K offset by +8.
// ---------------------------------------------------------------------------
DEV bf16x16 load_a_frag(const __bf16* lds, int lane, int mBase, int strideA) {
    const int m = mBase + (lane & 15);
    const int h = (lane >> 4) * 8;
    const __bf16* row = lds + m * strideA;
    bf16x16 a;
#pragma unroll
    for (int j = 0; j < 8; ++j) {
        const int kb = ((j < 4) ? 0 : 16) + h + (j & 3) * 2;
        a[2 * j]     = row[kb];
        a[2 * j + 1] = row[kb + 1];
    }
    return a;
}

// B: 32x16 bf16. N = lane&15; element s holds K in the same order as A
// (K = 8h+s for s<8, 16+8h+(s-8) for s>=8, h = lane>>4).
DEV bf16x16 load_b_frag(const __bf16* lds, int lane, int nBase, int strideB) {
    const int n = nBase + (lane & 15);
    const int h = (lane >> 4) * 8;
    bf16x16 b;
#pragma unroll
    for (int s = 0; s < 16; ++s) {
        const int k = ((s < 8) ? 0 : 16) + h + (s & 7);
        b[s] = lds[k * strideB + n];
    }
    return b;
}

// ---------------------------------------------------------------------------
// Tiled bf16 WMMA GEMM:  C[b] = scale * (A[b] @ B[b]) + bias
//   Block tile 128x128, 256 threads = 8 waves, each wave 32(M) x 64(N).
//   K-step 32. Double-buffered LDS; A (and non-transposed B) tiles staged with
//   CDNA5 GLOBAL_LOAD_ASYNC_TO_LDS_B128 so the k+1 fetch overlaps the k WMMAs.
//   BTRANS: B is stored [N,K] row-major (K^T for attention scores); its tile is
//   transposed through VGPRs into LDS (async copies cannot transpose 2B elems).
// ---------------------------------------------------------------------------
template <bool BTRANS, bool OUT_BF16>
__global__ __launch_bounds__(256) void gemm_wmma_bf16(
    const __bf16* __restrict__ Abase, long batchA, int lda,
    const __bf16* __restrict__ Bbase, long batchB, int ldb,
    void* __restrict__ Cbase, long batchC, int ldc,
    const float* __restrict__ bias, float scale, int K)
{
    constexpr int SA = 40;   // LDS A row stride (32 K + pad) -> 80B rows (16B aligned)
    constexpr int SB = 136;  // LDS B row stride (128 N + pad) -> 272B rows (16B aligned)
    __shared__ __bf16 ldsA[2][128 * SA];
    __shared__ __bf16 ldsB[2][32 * SB];

    const int t    = threadIdx.x;
    const int lane = t & 31;
    const int wave = t >> 5;
    const int wm   = wave >> 1;   // 0..3 -> 32-row band
    const int wn   = wave & 1;    // 0..1 -> 64-col band
    const long m0  = (long)blockIdx.y * 128;
    const long n0  = (long)blockIdx.x * 128;
    const int  bz  = blockIdx.z;

    const __bf16* A  = Abase + (long)bz * batchA;
    const __bf16* Bm = Bbase + (long)bz * batchB;

    v8f acc[2][4];
#pragma unroll
    for (int mi = 0; mi < 2; ++mi)
#pragma unroll
        for (int ni = 0; ni < 4; ++ni) {
            v8f z = {0.f, 0.f, 0.f, 0.f, 0.f, 0.f, 0.f, 0.f};
            acc[mi][ni] = z;
        }

    // staging work split (each thread moves 16 bf16 = 32B per tile)
    const int arow  = t >> 1, acol = (t & 1) * 16;   // A tile 128x32
    const int brow  = t >> 3, bcol = (t & 7) * 16;   // B tile 32x128 (normal)
    const int tnrow = t >> 1, tkc  = (t & 1) * 16;   // B tile 128x32 (trans)

    auto stage = [&](int k0, int buf) {
        { // A tile via async copy
            const __bf16* gsrc = A + (m0 + arow) * (long)lda + k0 + acol;
            __bf16* ldst = &ldsA[buf][arow * SA + acol];
            async_copy_b128(ldst,     gsrc);
            async_copy_b128(ldst + 8, gsrc + 8);
        }
        if constexpr (!BTRANS) { // B [K,N] tile via async copy
            const __bf16* gsrc = Bm + (long)(k0 + brow) * ldb + n0 + bcol;
            __bf16* ldst = &ldsB[buf][brow * SB + bcol];
            async_copy_b128(ldst,     gsrc);
            async_copy_b128(ldst + 8, gsrc + 8);
        } else {                 // B from [N,K] storage, transpose through VGPRs
            const __bf16* src = Bm + (n0 + tnrow) * (long)ldb + k0 + tkc;
            __bf16 tmp[16];
            *(uint4*)&tmp[0] = *(const uint4*)src;
            *(uint4*)&tmp[8] = *(const uint4*)(src + 8);
#pragma unroll
            for (int i = 0; i < 16; ++i)
                ldsB[buf][(tkc + i) * SB + tnrow] = tmp[i];
        }
    };

    // prologue: fill buffer 0
    stage(0, 0);
    wait_async_all();
    __syncthreads();

    int cur = 0;
    for (int k0 = 0; k0 < K; k0 += 32) {
        // issue next tile's fetch into the other buffer while we compute
        if (k0 + 32 < K) stage(k0 + 32, cur ^ 1);

        bf16x16 af[2], bfr[4];
#pragma unroll
        for (int mi = 0; mi < 2; ++mi)
            af[mi] = load_a_frag(ldsA[cur], lane, wm * 32 + mi * 16, SA);
#pragma unroll
        for (int ni = 0; ni < 4; ++ni)
            bfr[ni] = load_b_frag(ldsB[cur], lane, wn * 64 + ni * 16, SB);

#pragma unroll
        for (int mi = 0; mi < 2; ++mi)
#pragma unroll
            for (int ni = 0; ni < 4; ++ni)
                acc[mi][ni] = __builtin_amdgcn_wmma_f32_16x16x32_bf16(
                    false, af[mi], false, bfr[ni], (short)0, acc[mi][ni], false, false);

        // hand the freshly filled buffer to all waves
        wait_async_all();
        __syncthreads();
        cur ^= 1;
    }

    // epilogue: C/D layout -> VGPR j : lanes0-15 M=j, lanes16-31 M=j+8; N=lane&15
    float*  Cf = (float*)Cbase  + (long)bz * batchC;
    __bf16* Cb = (__bf16*)Cbase + (long)bz * batchC;
    const int r0 = (lane >> 4) * 8;
    const int cn = lane & 15;
#pragma unroll
    for (int mi = 0; mi < 2; ++mi)
#pragma unroll
        for (int ni = 0; ni < 4; ++ni) {
            const long rbase = m0 + wm * 32 + mi * 16 + r0;
            const long col   = n0 + wn * 64 + ni * 16 + cn;
            const float bvv  = bias ? bias[col] : 0.0f;
#pragma unroll
            for (int j = 0; j < 8; ++j) {
                const float v = acc[mi][ni][j] * scale + bvv;
                if constexpr (OUT_BF16)
                    Cb[(rbase + j) * (long)ldc + col] = (__bf16)v;
                else
                    Cf[(rbase + j) * (long)ldc + col] = v;
            }
        }
}

// ---------------------------------------------------------------------------
// fp32 -> bf16 (vectorized x4)
// ---------------------------------------------------------------------------
__global__ void cvt_f32_bf16_x4(const float* __restrict__ in,
                                __bf16* __restrict__ out, long n4)
{
    const long i = (long)blockIdx.x * blockDim.x + threadIdx.x;
    if (i >= n4) return;
    const float4 f = ((const float4*)in)[i];
    __bf16* o = out + i * 4;
    o[0] = (__bf16)f.x; o[1] = (__bf16)f.y; o[2] = (__bf16)f.z; o[3] = (__bf16)f.w;
}

// ---------------------------------------------------------------------------
// Row softmax over S elements; normalizes fp32 in-place + writes bf16 copy.
// ---------------------------------------------------------------------------
__global__ __launch_bounds__(256) void softmax_rows(float* __restrict__ sc,
                                                    __bf16* __restrict__ wbf, int S)
{
    const long row = blockIdx.x;
    float* p = sc + row * (long)S;
    __shared__ float red[256];
    const int t = threadIdx.x;

    float mx = -3.4e38f;
    for (int i = t; i < S; i += 256) mx = fmaxf(mx, p[i]);
    red[t] = mx; __syncthreads();
    for (int s = 128; s > 0; s >>= 1) { if (t < s) red[t] = fmaxf(red[t], red[t + s]); __syncthreads(); }
    mx = red[0]; __syncthreads();

    float sum = 0.f;
    for (int i = t; i < S; i += 256) { const float e = __expf(p[i] - mx); p[i] = e; sum += e; }
    red[t] = sum; __syncthreads();
    for (int s = 128; s > 0; s >>= 1) { if (t < s) red[t] += red[t + s]; __syncthreads(); }
    const float inv = 1.0f / red[0];

    __bf16* wb = wbf + row * (long)S;
    for (int i = t; i < S; i += 256) { const float wv = p[i] * inv; p[i] = wv; wb[i] = (__bf16)wv; }
}

// avg_w[b,k] = mean over q of weights[b,q,k]
__global__ void avg_weights(const float* __restrict__ wts, float* __restrict__ avg, int S)
{
    const int b = blockIdx.y;
    const int k = blockIdx.x * 256 + threadIdx.x;
    const float* p = wts + (long)b * S * S + k;
    float s = 0.f;
    for (int q = 0; q < S; ++q) s += p[(long)q * S];
    avg[b * S + k] = s * (1.0f / (float)S);
}

// y = LayerNorm(x + attn_out) * gamma + beta  (D = 1024)
__global__ __launch_bounds__(256) void residual_ln(const float* __restrict__ x,
                                                   const float* __restrict__ ao,
                                                   const float* __restrict__ gamma,
                                                   const float* __restrict__ beta,
                                                   float* __restrict__ y, int D)
{
    const long row = blockIdx.x;
    const int t = threadIdx.x;
    const float* xr = x  + row * (long)D;
    const float* ar = ao + row * (long)D;
    __shared__ float rs[256], rs2[256];
    float v[4], s = 0.f, s2 = 0.f;
#pragma unroll
    for (int i = 0; i < 4; ++i) {
        const int d = t + i * 256;
        const float val = xr[d] + ar[d];
        v[i] = val; s += val; s2 += val * val;
    }
    rs[t] = s; rs2[t] = s2; __syncthreads();
    for (int st = 128; st > 0; st >>= 1) {
        if (t < st) { rs[t] += rs[t + st]; rs2[t] += rs2[t + st]; }
        __syncthreads();
    }
    const float mean = rs[0] / (float)D;
    const float var  = rs2[0] / (float)D - mean * mean;
    const float inv  = rsqrtf(var + 1e-5f);
    float* yr = y + row * (long)D;
#pragma unroll
    for (int i = 0; i < 4; ++i) {
        const int d = t + i * 256;
        yr[d] = (v[i] - mean) * inv * gamma[d] + beta[d];
    }
}

// context[b,d] = sum_s avg_w[b,s] * y[b,s,d]
__global__ void context_reduce(const float* __restrict__ avg, const float* __restrict__ y,
                               float* __restrict__ ctx, int S, int D)
{
    const int b = blockIdx.y;
    const int d = blockIdx.x * 256 + threadIdx.x;
    const float* yb = y + (long)b * S * D + d;
    const float* ab = avg + b * S;
    float s = 0.f;
    for (int q = 0; q < S; ++q) s += ab[q] * yb[(long)q * D];
    ctx[b * D + d] = s;
}

// ---------------------------------------------------------------------------
// Host launcher
// ---------------------------------------------------------------------------
extern "C" void kernel_launch(void* const* d_in, const int* in_sizes, int n_in,
                              void* d_out, int out_size, void* d_ws, size_t ws_size,
                              hipStream_t stream)
{
    (void)in_sizes; (void)n_in; (void)out_size; (void)ws_size;

    const int  Bn = 8, S = 2048, Din = 1024, Datt = 1024;
    const long M  = (long)Bn * S;  // 16384

    const float* x     = (const float*)d_in[0];
    const float* Wq    = (const float*)d_in[1];
    const float* bq    = (const float*)d_in[2];
    const float* Wk    = (const float*)d_in[3];
    const float* bk    = (const float*)d_in[4];
    const float* Wv    = (const float*)d_in[5];
    const float* bv    = (const float*)d_in[6];
    const float* Wo    = (const float*)d_in[7];
    const float* bo    = (const float*)d_in[8];
    const float* gamma = (const float*)d_in[9];
    const float* beta  = (const float*)d_in[10];

    float* ctx_out = (float*)d_out;           // [8,1024]
    float* avg_out = ctx_out + Bn * Din;      // [8,2048]

    char* w = (char*)d_ws;
    size_t off = 0;
    auto alloc = [&](size_t bytes) -> void* {
        void* p = w + off;
        off = (off + bytes + 255) & ~(size_t)255;
        return p;
    };

    __bf16* xb     = (__bf16*)alloc(M * Din * 2);
    __bf16* Wqb    = (__bf16*)alloc((size_t)Din * Datt * 2);
    __bf16* Wkb    = (__bf16*)alloc((size_t)Din * Datt * 2);
    __bf16* Wvb    = (__bf16*)alloc((size_t)Din * Datt * 2);
    __bf16* Wob    = (__bf16*)alloc((size_t)Datt * Din * 2);
    __bf16* qb     = (__bf16*)alloc(M * Datt * 2);
    __bf16* kb     = (__bf16*)alloc(M * Datt * 2);
    __bf16* vb     = (__bf16*)alloc(M * Datt * 2);
    float*  scores = (float*)alloc((size_t)Bn * S * S * 4);
    __bf16* wbf    = (__bf16*)alloc((size_t)Bn * S * S * 2);
    __bf16* attnb  = (__bf16*)alloc(M * Datt * 2);
    float*  outF   = (float*)alloc(M * Din * 4);
    float*  y      = (float*)alloc(M * Din * 4);

    // 1) fp32 -> bf16 conversions
    {
        long n4 = M * Din / 4;
        cvt_f32_bf16_x4<<<dim3((unsigned)((n4 + 255) / 256)), 256, 0, stream>>>(x, xb, n4);
        long w4 = (long)Din * Datt / 4;
        dim3 g((unsigned)((w4 + 255) / 256));
        cvt_f32_bf16_x4<<<g, 256, 0, stream>>>(Wq, Wqb, w4);
        cvt_f32_bf16_x4<<<g, 256, 0, stream>>>(Wk, Wkb, w4);
        cvt_f32_bf16_x4<<<g, 256, 0, stream>>>(Wv, Wvb, w4);
        cvt_f32_bf16_x4<<<g, 256, 0, stream>>>(Wo, Wob, w4);
    }

    const float inv_sqrt_d = 0.03125f; // 1/sqrt(1024)

    // 2) Q/K/V projections: [16384,1024] = xb @ W + b   (bf16 out)
    {
        dim3 grid(Datt / 128, (unsigned)(M / 128), 1);
        gemm_wmma_bf16<false, true><<<grid, 256, 0, stream>>>(
            xb, 0, Din, Wqb, 0, Datt, qb, 0, Datt, bq, 1.0f, Din);
        gemm_wmma_bf16<false, true><<<grid, 256, 0, stream>>>(
            xb, 0, Din, Wkb, 0, Datt, kb, 0, Datt, bk, 1.0f, Din);
        gemm_wmma_bf16<false, true><<<grid, 256, 0, stream>>>(
            xb, 0, Din, Wvb, 0, Datt, vb, 0, Datt, bv, 1.0f, Din);
    }

    // 3) scores[b] = (q[b] @ k[b]^T) / sqrt(D)   (fp32 out, B transposed)
    {
        dim3 grid(S / 128, S / 128, Bn);
        gemm_wmma_bf16<true, false><<<grid, 256, 0, stream>>>(
            qb, (long)S * Datt, Datt,
            kb, (long)S * Datt, Datt,
            scores, (long)S * S, S,
            nullptr, inv_sqrt_d, Datt);
    }

    // 4) softmax rows (normalized fp32 in-place + bf16 copy)
    softmax_rows<<<dim3((unsigned)(Bn * S)), 256, 0, stream>>>(scores, wbf, S);

    // 5) avg_w = mean over q  -> straight into d_out
    avg_weights<<<dim3(S / 256, Bn), 256, 0, stream>>>(scores, avg_out, S);

    // 6) attn_out[b] = weights[b] @ v[b]   (bf16 out)
    {
        dim3 grid(Datt / 128, S / 128, Bn);
        gemm_wmma_bf16<false, true><<<grid, 256, 0, stream>>>(
            wbf, (long)S * S, S,
            vb,  (long)S * Datt, Datt,
            attnb, (long)S * Datt, Datt,
            nullptr, 1.0f, S);
    }

    // 7) out = attn_out @ Wo + bo   (fp32 out)
    {
        dim3 grid(Din / 128, (unsigned)(M / 128), 1);
        gemm_wmma_bf16<false, false><<<grid, 256, 0, stream>>>(
            attnb, 0, Datt, Wob, 0, Din, outF, 0, Din, bo, 1.0f, Datt);
    }

    // 8) y = LayerNorm(x + out)
    residual_ln<<<dim3((unsigned)M), 256, 0, stream>>>(x, outF, gamma, beta, y, Din);

    // 9) context = avg_w^T . y  -> d_out
    context_reduce<<<dim3(Din / 256, Bn), 256, 0, stream>>>(avg_out, y, ctx_out, S, Din);
}
